// NeighborhoodAttention1D_33878702031532
// MI455X (gfx1250) — matrix operand order repaired and checked by reading, not compiled, not run on hardware
//
#include <hip/hip_runtime.h>

// ---------------------------------------------------------------------------
// Types
// ---------------------------------------------------------------------------
typedef __attribute__((ext_vector_type(16))) __bf16        v16bf;
typedef __attribute__((ext_vector_type(8)))  float         v8f;
typedef __attribute__((ext_vector_type(4)))  float         f32x4;
typedef __attribute__((ext_vector_type(4)))  unsigned int  u32x4;
typedef __attribute__((ext_vector_type(2)))  unsigned int  u32x2;

union Frag {
    u32x4 u[2];   // 32 bytes
    v16bf b;      // 16 bf16
};

__device__ __forceinline__ unsigned short f2bf(float f) {
    unsigned int u = __float_as_uint(f);
    unsigned int r = u + 0x7FFFu + ((u >> 16) & 1u);   // round-to-nearest-even
    return (unsigned short)(r >> 16);
}
__device__ __forceinline__ unsigned int pack2(float a, float b) {
    return (unsigned int)f2bf(a) | ((unsigned int)f2bf(b) << 16);
}
__device__ __forceinline__ v8f zero8() {
    v8f z = {0.f, 0.f, 0.f, 0.f, 0.f, 0.f, 0.f, 0.f};
    return z;
}

// CDNA5 async copy: 32B per lane, global -> LDS (tracked by ASYNCcnt).
// INST_OFFSET applies to both the LDS and the global address.
__device__ __forceinline__ void async_copy_32B(unsigned lds_addr, const void* gaddr) {
    asm volatile("global_load_async_to_lds_b128 %0, %1, off\n\t"
                 "global_load_async_to_lds_b128 %0, %1, off offset:16"
                 :: "v"(lds_addr), "v"((unsigned long long)(uintptr_t)gaddr));
}
__device__ __forceinline__ void wait_asyncs() {
    asm volatile("s_wait_asynccnt 0x0" ::: "memory");
}

// Problem constants
#define BN   4
#define LL   4096
#define NH   16
#define HD   64
#define CC   1024
#define MTOT 16384          // B*L

// ---------------------------------------------------------------------------
// Weight transpose + fp32 -> bf16:  src (K x N, row major) -> dst (N x K)
// ---------------------------------------------------------------------------
__global__ void __launch_bounds__(256)
transpose_f32_to_bf16(const float* __restrict__ src, unsigned short* __restrict__ dst,
                      int K, int N)
{
    __shared__ float tile[32][33];
    const int n0 = blockIdx.x * 32;
    const int k0 = blockIdx.y * 32;
    const int t  = threadIdx.x;
    const int r  = t >> 3;          // 0..31
    const int c4 = (t & 7) * 4;     // 0..28

    f32x4 v = *(const f32x4*)(src + (size_t)(k0 + r) * N + n0 + c4);
    tile[r][c4 + 0] = v.x; tile[r][c4 + 1] = v.y;
    tile[r][c4 + 2] = v.z; tile[r][c4 + 3] = v.w;
    __syncthreads();

    const int n  = r;               // column within tile
    const int kk = c4;              // 4 consecutive k
    u32x2 o;
    o.x = pack2(tile[kk + 0][n], tile[kk + 1][n]);
    o.y = pack2(tile[kk + 2][n], tile[kk + 3][n]);
    *(u32x2*)(dst + (size_t)(n0 + n) * K + k0 + kk) = o;
}

// ---------------------------------------------------------------------------
// GEMM1: qkv = x @ w_qkv + b_qkv   (A fp32 on-the-fly -> bf16, B = wT bf16)
// Double-buffered LDS (offset-selected, keeps addrspace(3) inference);
// B staged via global_load_async_to_lds_b128.
// Epilogue scatters bf16 into q[b,h,l,d], k[b,h,l,d], vT[b,h,d,l]; q scaled.
// Block = 256 thr (8 waves, 2x4), block tile 128x128, K-step 32.
// Buffer i: A tile at i*20480, B tile at i*20480 + 10240 (80B padded rows).
// ---------------------------------------------------------------------------
__global__ void __launch_bounds__(256)
qkv_gemm(const float* __restrict__ X,            // 16384 x 1024
         const unsigned short* __restrict__ WT,  // 3072 x 1024 (bf16, transposed)
         const float* __restrict__ bias,         // 3072
         unsigned short* __restrict__ qb,
         unsigned short* __restrict__ kb,
         unsigned short* __restrict__ vTb)
{
    __shared__ __align__(16) char smem[40960];

    const int tid = threadIdx.x;
    const int nb = blockIdx.x, mb = blockIdx.y;
    const int w = tid >> 5, lane = tid & 31;
    const int wm = w >> 2, wn = w & 3;
    const int lhalf = lane >> 4, l16 = lane & 15;

    const int srow  = tid >> 1;     // 0..127
    const int shalf = tid & 1;      // 16-element half of 32-wide chunk
    const float*          Ag = X  + (size_t)(mb * 128 + srow) * CC + shalf * 16;
    const unsigned short* Bg = WT + (size_t)(nb * 128 + srow) * CC + shalf * 16;

    const unsigned stoff  = (unsigned)(srow * 80 + shalf * 32);   // staging offset in tile
    const unsigned sbase  = (unsigned)(uintptr_t)(void*)smem;     // LDS byte address of smem

    v8f acc[4][2];
#pragma unroll
    for (int i = 0; i < 4; i++)
#pragma unroll
        for (int j = 0; j < 2; j++) acc[i][j] = zero8();

    // prologue: stage K-step 0 into buffer 0
    {
        f32x4 a0 = *(const f32x4*)(Ag + 0);
        f32x4 a1 = *(const f32x4*)(Ag + 4);
        f32x4 a2 = *(const f32x4*)(Ag + 8);
        f32x4 a3 = *(const f32x4*)(Ag + 12);
        async_copy_32B(sbase + 10240u + stoff, Bg);
        u32x4 ap0 = {pack2(a0.x, a0.y), pack2(a0.z, a0.w), pack2(a1.x, a1.y), pack2(a1.z, a1.w)};
        u32x4 ap1 = {pack2(a2.x, a2.y), pack2(a2.z, a2.w), pack2(a3.x, a3.y), pack2(a3.z, a3.w)};
        *(u32x4*)(smem + stoff + 0)  = ap0;
        *(u32x4*)(smem + stoff + 16) = ap1;
        wait_asyncs();
    }
    __syncthreads();

#pragma unroll 2
    for (int step = 0; step < 32; ++step) {
        const unsigned curoff = (unsigned)(step & 1) * 20480u;
        const unsigned nxtoff = curoff ^ 20480u;
        const int kkn = (step + 1) * 32;
        const bool more = step < 31;

        f32x4 a0, a1, a2, a3;
        if (more) {
            a0 = *(const f32x4*)(Ag + kkn + 0);
            a1 = *(const f32x4*)(Ag + kkn + 4);
            a2 = *(const f32x4*)(Ag + kkn + 8);
            a3 = *(const f32x4*)(Ag + kkn + 12);
            async_copy_32B(sbase + nxtoff + 10240u + stoff, Bg + kkn);
        }

        // compute from current buffer (smem + integer offset -> ds_load)
        Frag af[4], bf[2];
#pragma unroll
        for (int mt = 0; mt < 4; mt++) {
            const char* p = smem + curoff + (wm * 64 + mt * 16 + l16) * 80 + lhalf * 16;
            af[mt].u[0] = *(const u32x4*)(p);
            af[mt].u[1] = *(const u32x4*)(p + 32);
        }
#pragma unroll
        for (int nt = 0; nt < 2; nt++) {
            const char* p = smem + curoff + 10240 + (wn * 32 + nt * 16 + l16) * 80 + lhalf * 16;
            bf[nt].u[0] = *(const u32x4*)(p);
            bf[nt].u[1] = *(const u32x4*)(p + 32);
        }
#pragma unroll
        for (int mt = 0; mt < 4; mt++)
#pragma unroll
            for (int nt = 0; nt < 2; nt++)
                acc[mt][nt] = __builtin_amdgcn_wmma_f32_16x16x32_bf16(
                    false, af[mt].b, false, bf[nt].b, (short)0, acc[mt][nt], false, false);

        if (more) {
            u32x4 ap0 = {pack2(a0.x, a0.y), pack2(a0.z, a0.w), pack2(a1.x, a1.y), pack2(a1.z, a1.w)};
            u32x4 ap1 = {pack2(a2.x, a2.y), pack2(a2.z, a2.w), pack2(a3.x, a3.y), pack2(a3.z, a3.w)};
            *(u32x4*)(smem + nxtoff + stoff + 0)  = ap0;
            *(u32x4*)(smem + nxtoff + stoff + 16) = ap1;
        }
        wait_asyncs();
        __syncthreads();
    }

    // epilogue: bias, scale q, scatter
#pragma unroll
    for (int nt = 0; nt < 2; nt++) {
        const int n = nb * 128 + wn * 32 + nt * 16 + l16;
        const int which = n >> 10;
        const int h = (n >> 6) & 15;
        const int d = n & 63;
        const float bn = bias[n];
        unsigned short* base = (which == 0) ? qb : (which == 1) ? kb : vTb;
#pragma unroll
        for (int mt = 0; mt < 4; mt++) {
#pragma unroll
            for (int r = 0; r < 8; r++) {
                const int m = mb * 128 + wm * 64 + mt * 16 + r + lhalf * 8;
                const int b_ = m >> 12;
                const int l  = m & 4095;
                float val = acc[mt][nt][r] + bn;
                if (which == 0) val *= 0.125f;   // HEAD_DIM^-0.5
                size_t off;
                if (which == 2) off = ((size_t)((b_ * NH + h) * HD + d)) * LL + l;
                else            off = ((size_t)((b_ * NH + h) * LL + l)) * HD + d;
                base[off] = f2bf(val);
            }
        }
    }
}

// ---------------------------------------------------------------------------
// Neighborhood attention: 1 wave per 16-query tile, WMMA for S and O.
// ---------------------------------------------------------------------------
__global__ void __launch_bounds__(256)
natten_kernel(const unsigned short* __restrict__ q,   // [B,H,L,64]
              const unsigned short* __restrict__ k,   // [B,H,L,64]
              const unsigned short* __restrict__ vT,  // [B,H,64,L]
              const float* __restrict__ rpb,          // [16,25]
              unsigned short* __restrict__ o)         // [B,L,1024] bf16
{
    __shared__ __align__(16) char pmem[8 * 16 * 80];   // per-wave P: 16 rows x 80B
    const int tid = threadIdx.x;
    const int w = tid >> 5, lane = tid & 31;
    const int lhalf = lane >> 4, l16 = lane & 15;

    const int tile = blockIdx.x * 8 + w;
    const int b = tile >> 12;
    const int h = (tile >> 8) & 15;
    const int l0 = (tile & 255) * 16;

    int ks = l0 - 8;                       // multiple of 8 -> aligned slab
    if (ks < 0) ks = 0;
    if (ks > LL - 32) ks = LL - 32;

    const unsigned short* qp = q  + (size_t)(b * NH + h) * LL * HD;
    const unsigned short* kp = k  + (size_t)(b * NH + h) * LL * HD;
    const unsigned short* vp = vT + (size_t)(b * NH + h) * HD * LL;

    // Q A-fragments (two 32-d chunks)
    Frag qa[2];
#pragma unroll
    for (int c = 0; c < 2; c++) {
        const unsigned short* p = qp + (size_t)(l0 + l16) * HD + c * 32 + lhalf * 8;
        qa[c].u[0] = *(const u32x4*)(p);
        qa[c].u[1] = *(const u32x4*)(p + 16);
    }

    // S = Q @ K^T  (16 x 32 keys)
    v8f s0 = zero8(), s1 = zero8();
#pragma unroll
    for (int c = 0; c < 2; c++) {
#pragma unroll
        for (int nt = 0; nt < 2; nt++) {
            Frag kf;
            const unsigned short* p = kp + (size_t)(ks + nt * 16 + l16) * HD + c * 32 + lhalf * 8;
            kf.u[0] = *(const u32x4*)(p);
            kf.u[1] = *(const u32x4*)(p + 16);
            if (nt == 0) s0 = __builtin_amdgcn_wmma_f32_16x16x32_bf16(false, qa[c].b, false, kf.b, (short)0, s0, false, false);
            else         s1 = __builtin_amdgcn_wmma_f32_16x16x32_bf16(false, qa[c].b, false, kf.b, (short)0, s1, false, false);
        }
    }

    // masked softmax with relative position bias (rows split across lane halves)
#pragma unroll
    for (int r = 0; r < 8; r++) {
        const int i  = l0 + r + lhalf * 8;
        int ni = i - 6;
        if (ni < 0) ni = 0;
        if (ni > LL - 13) ni = LL - 13;
        const int key0 = ks + l16;
        const int key1 = ks + 16 + l16;
        const bool v0 = (key0 >= ni) && (key0 <= ni + 12);
        const bool v1 = (key1 >= ni) && (key1 <= ni + 12);
        float e0 = v0 ? s0[r] + rpb[h * 25 + key0 - i + 12] : -3.0e38f;
        float e1 = v1 ? s1[r] + rpb[h * 25 + key1 - i + 12] : -3.0e38f;
        float mx = fmaxf(e0, e1);
#pragma unroll
        for (int dlt = 1; dlt < 16; dlt <<= 1) mx = fmaxf(mx, __shfl_xor(mx, dlt, 32));
        float p0 = v0 ? __expf(e0 - mx) : 0.f;
        float p1 = v1 ? __expf(e1 - mx) : 0.f;
        float sum = p0 + p1;
#pragma unroll
        for (int dlt = 1; dlt < 16; dlt <<= 1) sum += __shfl_xor(sum, dlt, 32);
        const float rs = 1.f / sum;
        s0[r] = p0 * rs;
        s1[r] = p1 * rs;
    }

    // P (D-layout) -> LDS bf16 -> A-fragment layout
    unsigned short* pl = (unsigned short*)(pmem + w * 16 * 80);
#pragma unroll
    for (int r = 0; r < 8; r++) {
        const int row = r + lhalf * 8;
        pl[row * 40 + l16]      = f2bf(s0[r]);
        pl[row * 40 + 16 + l16] = f2bf(s1[r]);
    }
    __syncthreads();

    Frag pf;
    {
        const unsigned short* pp = pl + l16 * 40 + lhalf * 8;
        pf.u[0] = *(const u32x4*)(pp);
        pf.u[1] = *(const u32x4*)(pp + 16);
    }

    // O = P @ V  (V^T rows are key-contiguous -> direct B-fragments)
    v8f oacc[4];
#pragma unroll
    for (int nt = 0; nt < 4; nt++) oacc[nt] = zero8();
#pragma unroll
    for (int nt = 0; nt < 4; nt++) {
        Frag vf;
        const unsigned short* p = vp + (size_t)(nt * 16 + l16) * LL + ks + lhalf * 8;
        vf.u[0] = *(const u32x4*)(p);
        vf.u[1] = *(const u32x4*)(p + 16);
        oacc[nt] = __builtin_amdgcn_wmma_f32_16x16x32_bf16(false, pf.b, false, vf.b, (short)0, oacc[nt], false, false);
    }

    // store bf16 out [b, l, h*64 + d]
#pragma unroll
    for (int nt = 0; nt < 4; nt++) {
        const int d = nt * 16 + l16;
#pragma unroll
        for (int r = 0; r < 8; r++) {
            const int l = l0 + r + lhalf * 8;
            o[((size_t)(b * LL + l)) * CC + h * HD + d] = f2bf(oacc[nt][r]);
        }
    }
}

// ---------------------------------------------------------------------------
// GEMM2: out = attn @ w_proj + b_proj   (A bf16, B = wT bf16, out fp32)
// Double-buffered LDS; both A and B staged via async-to-LDS.
// ---------------------------------------------------------------------------
__global__ void __launch_bounds__(256)
proj_gemm(const unsigned short* __restrict__ A,      // 16384 x 1024 bf16
          const unsigned short* __restrict__ WT,     // 1024 x 1024 bf16 (transposed)
          const float* __restrict__ bias,            // 1024
          float* __restrict__ out)                   // 16384 x 1024
{
    __shared__ __align__(16) char smem[40960];

    const int tid = threadIdx.x;
    const int nb = blockIdx.x, mb = blockIdx.y;
    const int w = tid >> 5, lane = tid & 31;
    const int wm = w >> 2, wn = w & 3;
    const int lhalf = lane >> 4, l16 = lane & 15;

    const int srow  = tid >> 1;
    const int shalf = tid & 1;
    const unsigned short* Ag = A  + (size_t)(mb * 128 + srow) * CC + shalf * 16;
    const unsigned short* Bg = WT + (size_t)(nb * 128 + srow) * CC + shalf * 16;

    const unsigned stoff = (unsigned)(srow * 80 + shalf * 32);
    const unsigned sbase = (unsigned)(uintptr_t)(void*)smem;

    v8f acc[4][2];
#pragma unroll
    for (int i = 0; i < 4; i++)
#pragma unroll
        for (int j = 0; j < 2; j++) acc[i][j] = zero8();

    // prologue: stage K-step 0 into buffer 0 (fully async)
    async_copy_32B(sbase + stoff, Ag);
    async_copy_32B(sbase + 10240u + stoff, Bg);
    wait_asyncs();
    __syncthreads();

#pragma unroll 2
    for (int step = 0; step < 32; ++step) {
        const unsigned curoff = (unsigned)(step & 1) * 20480u;
        const unsigned nxtoff = curoff ^ 20480u;
        const int kkn = (step + 1) * 32;
        const bool more = step < 31;

        if (more) {
            async_copy_32B(sbase + nxtoff + stoff, Ag + kkn);
            async_copy_32B(sbase + nxtoff + 10240u + stoff, Bg + kkn);
        }

        Frag af[4], bf[2];
#pragma unroll
        for (int mt = 0; mt < 4; mt++) {
            const char* p = smem + curoff + (wm * 64 + mt * 16 + l16) * 80 + lhalf * 16;
            af[mt].u[0] = *(const u32x4*)(p);
            af[mt].u[1] = *(const u32x4*)(p + 32);
        }
#pragma unroll
        for (int nt = 0; nt < 2; nt++) {
            const char* p = smem + curoff + 10240 + (wn * 32 + nt * 16 + l16) * 80 + lhalf * 16;
            bf[nt].u[0] = *(const u32x4*)(p);
            bf[nt].u[1] = *(const u32x4*)(p + 32);
        }
#pragma unroll
        for (int mt = 0; mt < 4; mt++)
#pragma unroll
            for (int nt = 0; nt < 2; nt++)
                acc[mt][nt] = __builtin_amdgcn_wmma_f32_16x16x32_bf16(
                    false, af[mt].b, false, bf[nt].b, (short)0, acc[mt][nt], false, false);

        wait_asyncs();
        __syncthreads();
    }

#pragma unroll
    for (int nt = 0; nt < 2; nt++) {
        const int n = nb * 128 + wn * 32 + nt * 16 + l16;
        const float bn = bias[n];
#pragma unroll
        for (int mt = 0; mt < 4; mt++) {
#pragma unroll
            for (int r = 0; r < 8; r++) {
                const int m = mb * 128 + wm * 64 + mt * 16 + r + lhalf * 8;
                out[(size_t)m * CC + n] = acc[mt][nt][r] + bn;
            }
        }
    }
}

// ---------------------------------------------------------------------------
// Launch
// ---------------------------------------------------------------------------
extern "C" void kernel_launch(void* const* d_in, const int* in_sizes, int n_in,
                              void* d_out, int out_size, void* d_ws, size_t ws_size,
                              hipStream_t stream)
{
    (void)in_sizes; (void)n_in; (void)out_size; (void)ws_size;
    const float* x      = (const float*)d_in[0];
    const float* w_qkv  = (const float*)d_in[1];
    const float* b_qkv  = (const float*)d_in[2];
    const float* rpb    = (const float*)d_in[3];
    const float* w_proj = (const float*)d_in[4];
    const float* b_proj = (const float*)d_in[5];
    float* out = (float*)d_out;

    char* ws = (char*)d_ws;
    // workspace layout (bytes)
    unsigned short* wqkvT  = (unsigned short*)(ws);                              // 3072*1024*2 = 6291456
    unsigned short* wprojT = (unsigned short*)(ws + 6291456);                    // 1024*1024*2 = 2097152
    unsigned short* qbuf   = (unsigned short*)(ws + 8388608);                    // 33554432
    unsigned short* kbuf   = (unsigned short*)(ws + 8388608 + 33554432);         // 33554432
    unsigned short* vTbuf  = (unsigned short*)(ws + 8388608 + 2u * 33554432u);   // 33554432
    unsigned short* attn   = (unsigned short*)(ws + 8388608 + 3u * 33554432u);   // 33554432

    transpose_f32_to_bf16<<<dim3(96, 32), 256, 0, stream>>>(w_qkv, wqkvT, 1024, 3072);
    transpose_f32_to_bf16<<<dim3(32, 32), 256, 0, stream>>>(w_proj, wprojT, 1024, 1024);
    qkv_gemm<<<dim3(24, 128), 256, 0, stream>>>(x, wqkvT, b_qkv, qbuf, kbuf, vTbuf);
    natten_kernel<<<dim3(2048), 256, 0, stream>>>(qbuf, kbuf, vTbuf, rpb, attn);
    proj_gemm<<<dim3(8, 128), 256, 0, stream>>>(attn, wprojT, b_proj, out);
}
